// MultiHeadedAttention_32762010534471
// MI455X (gfx1250) — matrix operand order repaired and checked by reading, not compiled, not run on hardware
//
#include <hip/hip_runtime.h>

// MHA forward for MI455X (gfx1250, wave32).
// bf16 WMMA everywhere, f32 accumulate; flash-attention (no SxS score tensor).
// Layouts: Q,K as [B*H][S][64] bf16; V as [B*H][64][S] bf16 (pre-transposed so
// V tiles load directly as WMMA B-fragments); ctx as [B,S,1024] bf16.
// GEMMs use async global->LDS (ASYNCcnt) double buffering when the toolchain
// exposes the gfx1250 async builtins; otherwise register double buffering.

#define HEADS   16
#define HS      64
#define HIDDEN  1024
#define BATCH   4
#define SEQ     2048
#define MTOT    (BATCH * SEQ)   // 8192 rows for projection GEMMs

#if defined(__has_builtin)
# if __has_builtin(__builtin_amdgcn_global_load_async_to_lds_b128)
#   define HAVE_ASYNC_LDS 1
# endif
# if __has_builtin(__builtin_amdgcn_s_wait_asynccnt)
#   define HAVE_WAIT_ASYNC 1
# endif
#endif
#ifndef HAVE_ASYNC_LDS
# define HAVE_ASYNC_LDS 0
#endif
#ifndef HAVE_WAIT_ASYNC
# define HAVE_WAIT_ASYNC 0
#endif

typedef __attribute__((ext_vector_type(16))) __bf16 v16bf;
typedef __attribute__((ext_vector_type(8)))  float  v8f;

union FragB16 { v16bf v; uint4 u[2]; };
union FragC   { v8f   v; float f[8]; };

// ---- fragment loaders (wave32 WMMA 16x16x32 bf16 layouts) -------------------
// A (16x32, MxK): lane m = lane&15 holds row m; half = lane>>4.
//   dwords 0-3 = K[8*half .. 8*half+7], dwords 4-7 = K[8*half+16 .. +23]
__device__ __forceinline__ v16bf frag_a(const __bf16* base, int stride, int lane) {
  const int m = lane & 15, h = lane >> 4;
  const __bf16* p = base + (size_t)m * stride + h * 8;
  FragB16 f;
  f.u[0] = *(const uint4*)(p);
  f.u[1] = *(const uint4*)(p + 16);
  return f.v;
}
// B (32x16, KxN) loaded from a row-major B^T tile [N][K]:
//   lane n = lane&15 holds column n; same K order as A: K[8*half..+7], K[8*half+16..+23].
__device__ __forceinline__ v16bf frag_b(const __bf16* base, size_t stride, int lane) {
  const int n = lane & 15, h = lane >> 4;
  const __bf16* p = base + (size_t)n * stride + h * 8;
  FragB16 f;
  f.u[0] = *(const uint4*)(p);
  f.u[1] = *(const uint4*)(p + 16);
  return f.v;
}
__device__ __forceinline__ v8f wmma_bf16(v16bf a, v16bf b, v8f c) {
  return __builtin_amdgcn_wmma_f32_16x16x32_bf16(false, a, false, b, (short)0, c,
                                                 false, false);
}

#if HAVE_ASYNC_LDS
// Prototype (from compiler diagnostic): first param is v4i32 in the global
// address space; LDS side is the AS(3) counterpart. Explicit addrspace casts
// are legal because generic overlaps both.
typedef int v4i32 __attribute__((__vector_size__(4 * sizeof(int))));
typedef __attribute__((address_space(1))) v4i32* gbl_v4i32_ptr;
typedef __attribute__((address_space(3))) v4i32* lds_v4i32_ptr;

__device__ __forceinline__ void async_b128(const __bf16* g, __bf16* l) {
  __builtin_amdgcn_global_load_async_to_lds_b128(
      (gbl_v4i32_ptr)(g), (lds_v4i32_ptr)(l), 0, 0);
}
__device__ __forceinline__ void wait_async0() {
#if HAVE_WAIT_ASYNC
  __builtin_amdgcn_s_wait_asynccnt(0);
#else
  asm volatile("s_wait_asynccnt 0" ::: "memory");
#endif
}
#endif

// ---- f32 -> bf16 elementwise convert ---------------------------------------
__global__ __launch_bounds__(256)
void convert_f32_bf16(const float* __restrict__ in, __bf16* __restrict__ out, int n) {
  int i = (blockIdx.x * 256 + threadIdx.x) * 8;
  if (i + 8 > n) return;
  float4 a = *(const float4*)(in + i);
  float4 b = *(const float4*)(in + i + 4);
  union { __bf16 h[8]; uint4 u; } o;
  o.h[0] = (__bf16)a.x; o.h[1] = (__bf16)a.y; o.h[2] = (__bf16)a.z; o.h[3] = (__bf16)a.w;
  o.h[4] = (__bf16)b.x; o.h[5] = (__bf16)b.y; o.h[6] = (__bf16)b.z; o.h[7] = (__bf16)b.w;
  *(uint4*)(out + i) = o.u;
}

// ---- W [K][N] f32 -> W^T [N][K] bf16 (LDS tiled transpose) ------------------
__global__ __launch_bounds__(256)
void transpose_w(const float* __restrict__ W, __bf16* __restrict__ Wt) {
  __shared__ float tile[32][33];
  const int bn = blockIdx.x * 32;   // N block
  const int bk = blockIdx.y * 32;   // K block
  const int tx = threadIdx.x & 31, ty = threadIdx.x >> 5;
  for (int i = ty; i < 32; i += 8)
    tile[i][tx] = W[(size_t)(bk + i) * HIDDEN + bn + tx];
  __syncthreads();
  for (int i = ty; i < 32; i += 8)
    Wt[(size_t)(bn + i) * HIDDEN + bk + tx] = (__bf16)tile[tx][i];
}

// ---- bf16 GEMM: Y = (X @ W + bias) * scale ---------------------------------
// X: [MTOT][1024] bf16 row-major, Wt: [1024][1024] bf16 (W transposed).
// MODE 0: outf f32 [MTOT][1024]     (final projection)
// MODE 1: outb bf16 [B*H][S][64]    (Q with scale=1/8, K)
// MODE 2: outb bf16 [B*H][64][S]    (V transposed)
template <int MODE>
__global__ __launch_bounds__(256)
void gemm_bf16_kernel(const __bf16* __restrict__ X, const __bf16* __restrict__ Wt,
                      const float* __restrict__ bias, float scale,
                      __bf16* __restrict__ outb, float* __restrict__ outf) {
  __shared__ __bf16 As[2][128 * 32];   // double-buffered: 128 rows x 32 K
  __shared__ __bf16 Bs[2][64 * 32];    // double-buffered: 64 B^T rows x 32 K
  const int tid  = threadIdx.x;
  const int lane = tid & 31, wid = tid >> 5;
  const int wm = wid >> 1, wn = wid & 1;            // 4x2 wave grid
  const int m0 = blockIdx.y * 128, n0 = blockIdx.x * 64;

  // staging assignment: A -> 32B/thread (2 x b128), B^T -> 16B/thread (1 x b128)
  const int arow = tid >> 1, ah = (tid & 1) * 16;
  const int brow = tid >> 2, bq = (tid & 3) * 8;
  const __bf16* ag = X  + (size_t)(m0 + arow) * HIDDEN + ah;
  const __bf16* bg = Wt + (size_t)(n0 + brow) * HIDDEN + bq;

  v8f acc[2][2] = {};

#if HAVE_ASYNC_LDS
  // ---- async global->LDS double buffer (ASYNCcnt) ----
  {
    async_b128(ag, &As[0][arow * 32 + ah]);
    async_b128(ag + 8, &As[0][arow * 32 + ah + 8]);
    async_b128(bg, &Bs[0][brow * 32 + bq]);
  }
  for (int k0 = 0; k0 < HIDDEN; k0 += 32) {
    const int cur = (k0 >> 5) & 1;
    wait_async0();        // my async writes done...
    __syncthreads();      // ...and everyone else's too
    if (k0 + 32 < HIDDEN) {
      const int nxt = cur ^ 1;
      async_b128(ag + k0 + 32, &As[nxt][arow * 32 + ah]);
      async_b128(ag + k0 + 40, &As[nxt][arow * 32 + ah + 8]);
      async_b128(bg + k0 + 32, &Bs[nxt][brow * 32 + bq]);
    }
    v16bf a0 = frag_a(&As[cur][(wm * 32 + 0) * 32], 32, lane);
    v16bf a1 = frag_a(&As[cur][(wm * 32 + 16) * 32], 32, lane);
    v16bf b0 = frag_b(&Bs[cur][(wn * 32 + 0) * 32], 32, lane);
    v16bf b1 = frag_b(&Bs[cur][(wn * 32 + 16) * 32], 32, lane);
    acc[0][0] = wmma_bf16(a0, b0, acc[0][0]);
    acc[0][1] = wmma_bf16(a0, b1, acc[0][1]);
    acc[1][0] = wmma_bf16(a1, b0, acc[1][0]);
    acc[1][1] = wmma_bf16(a1, b1, acc[1][1]);
  }
#else
  // ---- register-staged double buffer (one barrier per K step) ----
  uint4 ra0, ra1, rb;
  ra0 = *(const uint4*)(ag);
  ra1 = *(const uint4*)(ag + 8);
  rb  = *(const uint4*)(bg);
  for (int k0 = 0; k0 < HIDDEN; k0 += 32) {
    const int cur = (k0 >> 5) & 1;
    uint4* la = (uint4*)&As[cur][arow * 32 + ah];
    la[0] = ra0;
    la[1] = ra1;
    *(uint4*)&Bs[cur][brow * 32 + bq] = rb;
    if (k0 + 64 < HIDDEN)   // warm L2/WGP$ two tiles ahead (global_prefetch)
      __builtin_prefetch(ag + k0 + 64, 0, 1);
    __syncthreads();
    if (k0 + 32 < HIDDEN) {
      ra0 = *(const uint4*)(ag + k0 + 32);
      ra1 = *(const uint4*)(ag + k0 + 40);
      rb  = *(const uint4*)(bg + k0 + 32);
    }
    v16bf a0 = frag_a(&As[cur][(wm * 32 + 0) * 32], 32, lane);
    v16bf a1 = frag_a(&As[cur][(wm * 32 + 16) * 32], 32, lane);
    v16bf b0 = frag_b(&Bs[cur][(wn * 32 + 0) * 32], 32, lane);
    v16bf b1 = frag_b(&Bs[cur][(wn * 32 + 16) * 32], 32, lane);
    acc[0][0] = wmma_bf16(a0, b0, acc[0][0]);
    acc[0][1] = wmma_bf16(a0, b1, acc[0][1]);
    acc[1][0] = wmma_bf16(a1, b0, acc[1][0]);
    acc[1][1] = wmma_bf16(a1, b1, acc[1][1]);
  }
#endif

  // epilogue: C layout -> row = r + 8*(lane>>4), col = lane&15
  const int half = lane >> 4, nl = lane & 15;
  for (int i = 0; i < 2; ++i)
    for (int j = 0; j < 2; ++j) {
      FragC c; c.v = acc[i][j];
      for (int r = 0; r < 8; ++r) {
        const int gm = m0 + wm * 32 + i * 16 + r + 8 * half;
        const int gn = n0 + wn * 32 + j * 16 + nl;
        const float y = (c.f[r] + bias[gn]) * scale;
        if (MODE == 0) {
          outf[(size_t)gm * HIDDEN + gn] = y;
        } else {
          const int b = gm >> 11, s = gm & (SEQ - 1);   // S = 2048
          const int hh = gn >> 6, d = gn & 63;          // head size 64
          if (MODE == 1)
            outb[(((size_t)(b * HEADS + hh)) * SEQ + s) * HS + d] = (__bf16)y;
          else
            outb[(((size_t)(b * HEADS + hh)) * HS + d) * SEQ + s] = (__bf16)y;
        }
      }
    }
}

// ---- flash attention: each wave owns a 16-row Q strip ----------------------
__global__ __launch_bounds__(256)
void attn_kernel(const __bf16* __restrict__ Q, const __bf16* __restrict__ Kb,
                 const __bf16* __restrict__ Vt, const unsigned char* __restrict__ mask,
                 __bf16* __restrict__ ctx) {
  __shared__ __bf16 Plds[8][16 * 32];   // per-wave prob staging (C-layout -> A-layout)
  const int lane = threadIdx.x & 31, wid = threadIdx.x >> 5;
  const int half = lane >> 4, nl = lane & 15;
  const int bh = blockIdx.y;
  const int b = bh >> 4, hh = bh & 15;
  const int s0 = blockIdx.x * 128 + wid * 16;

  const __bf16* Qbase = Q + ((size_t)bh * SEQ + s0) * HS;
  const v16bf qa0 = frag_a(Qbase, HS, lane);        // d 0..31
  const v16bf qa1 = frag_a(Qbase + 32, HS, lane);   // d 32..63

  FragC o[4] = {};                                  // 16x64 ctx accumulator
  float mrow[8], lrow[8];
  for (int r = 0; r < 8; ++r) { mrow[r] = -1e30f; lrow[r] = 0.0f; }

  const __bf16* Kbh = Kb + (size_t)bh * SEQ * HS;
  const __bf16* Vbh = Vt + (size_t)bh * HS * SEQ;
  const unsigned char* Mb = mask + (size_t)b * SEQ * SEQ;
  __bf16* P = &Plds[wid][0];

  for (int kb = 0; kb < SEQ; kb += 32) {
    // scores: two 16x16 tiles covering 32 key columns
    FragC sc[2];
    for (int t = 0; t < 2; ++t) {
      const __bf16* Kt = Kbh + (size_t)(kb + 16 * t) * HS;
      v16bf kf0 = frag_b(Kt, HS, lane);        // K rows as B^T, d 0..31
      v16bf kf1 = frag_b(Kt + 32, HS, lane);   // d 32..63
      v8f s = {};
      s = wmma_bf16(qa0, kf0, s);
      s = wmma_bf16(qa1, kf1, s);
      sc[t].v = s;
    }
    // mask (true -> -1e18), coalesced byte loads
    for (int r = 0; r < 8; ++r) {
      const unsigned char* mp = Mb + (size_t)(s0 + r + 8 * half) * SEQ + kb;
      if (mp[nl])      sc[0].f[r] = -1e18f;
      if (mp[16 + nl]) sc[1].f[r] = -1e18f;
    }
    // online softmax; rows live in 16-lane halves -> reduce with xor 1,2,4,8
    for (int r = 0; r < 8; ++r) {
      float v = fmaxf(sc[0].f[r], sc[1].f[r]);
      v = fmaxf(v, __shfl_xor(v, 1, 32));
      v = fmaxf(v, __shfl_xor(v, 2, 32));
      v = fmaxf(v, __shfl_xor(v, 4, 32));
      v = fmaxf(v, __shfl_xor(v, 8, 32));
      const float mnew  = fmaxf(mrow[r], v);
      const float alpha = __expf(mrow[r] - mnew);
      mrow[r] = mnew;
      const float p0 = __expf(sc[0].f[r] - mnew);
      const float p1 = __expf(sc[1].f[r] - mnew);
      float ps = p0 + p1;
      ps += __shfl_xor(ps, 1, 32);
      ps += __shfl_xor(ps, 2, 32);
      ps += __shfl_xor(ps, 4, 32);
      ps += __shfl_xor(ps, 8, 32);
      lrow[r] = lrow[r] * alpha + ps;
      o[0].f[r] *= alpha; o[1].f[r] *= alpha;
      o[2].f[r] *= alpha; o[3].f[r] *= alpha;
      // stash probs (C-layout element -> its [row][col] slot)
      P[(r + 8 * half) * 32 + nl]      = (__bf16)p0;
      P[(r + 8 * half) * 32 + 16 + nl] = (__bf16)p1;
    }
    // wave-private LDS round-trip: wait DS writes, reload as A-fragment
    asm volatile("s_wait_dscnt 0" ::: "memory");
    const v16bf pa = frag_a(P, 32, lane);
    for (int t2 = 0; t2 < 4; ++t2) {
      v16bf vb = frag_b(Vbh + (size_t)(16 * t2) * SEQ + kb, (size_t)SEQ, lane);
      o[t2].v = wmma_bf16(pa, vb, o[t2].v);
    }
    asm volatile("s_wait_dscnt 0" ::: "memory");  // reads done before next-iter writes
  }

  for (int r = 0; r < 8; ++r) {
    const float inv = 1.0f / lrow[r];
    const int row = s0 + r + 8 * half;
    const size_t base = ((size_t)b * SEQ + row) * HIDDEN + hh * HS;
    for (int t2 = 0; t2 < 4; ++t2)
      ctx[base + 16 * t2 + nl] = (__bf16)(o[t2].f[r] * inv);
  }
}

// ---- host-side orchestration ------------------------------------------------
extern "C" void kernel_launch(void* const* d_in, const int* in_sizes, int n_in,
                              void* d_out, int out_size, void* d_ws, size_t ws_size,
                              hipStream_t stream) {
  const float* key   = (const float*)d_in[0];
  const float* value = (const float*)d_in[1];
  const float* query = (const float*)d_in[2];
  const unsigned char* mask = (const unsigned char*)d_in[3];  // jnp bool = 1 byte
  const float* Wq = (const float*)d_in[4];
  const float* bq = (const float*)d_in[5];
  const float* Wk = (const float*)d_in[6];
  const float* bk = (const float*)d_in[7];
  const float* Wv = (const float*)d_in[8];
  const float* bv = (const float*)d_in[9];
  const float* Wo = (const float*)d_in[10];
  const float* bo = (const float*)d_in[11];

  __bf16* ws = (__bf16*)d_ws;
  const size_t ACT = (size_t)MTOT * HIDDEN;    // 8M elems
  const size_t WEL = (size_t)HIDDEN * HIDDEN;  // 1M elems
  __bf16* Xq  = ws;             // converted query
  __bf16* Xk  = Xq  + ACT;      // converted key
  __bf16* Xv  = Xk  + ACT;      // converted value
  __bf16* Wqt = Xv  + ACT;      // transposed bf16 weights
  __bf16* Wkt = Wqt + WEL;
  __bf16* Wvt = Wkt + WEL;
  __bf16* Wot = Wvt + WEL;
  __bf16* Qb  = Wot + WEL;      // [B*H][S][64]
  __bf16* Kh  = Qb  + ACT;      // [B*H][S][64]
  __bf16* Vt  = Kh  + ACT;      // [B*H][64][S]
  __bf16* Cx  = Vt  + ACT;      // [B][S][1024]

  const dim3 cvtg((unsigned)(ACT / (256 * 8)));
  convert_f32_bf16<<<cvtg, 256, 0, stream>>>(query, Xq, (int)ACT);
  convert_f32_bf16<<<cvtg, 256, 0, stream>>>(key,   Xk, (int)ACT);
  convert_f32_bf16<<<cvtg, 256, 0, stream>>>(value, Xv, (int)ACT);

  const dim3 tg(HIDDEN / 32, HIDDEN / 32);
  transpose_w<<<tg, 256, 0, stream>>>(Wq, Wqt);
  transpose_w<<<tg, 256, 0, stream>>>(Wk, Wkt);
  transpose_w<<<tg, 256, 0, stream>>>(Wv, Wvt);
  transpose_w<<<tg, 256, 0, stream>>>(Wo, Wot);

  const dim3 gg(HIDDEN / 64, MTOT / 128);
  gemm_bf16_kernel<1><<<gg, 256, 0, stream>>>(Xq, Wqt, bq, 0.125f, Qb, nullptr); // /sqrt(64)
  gemm_bf16_kernel<1><<<gg, 256, 0, stream>>>(Xk, Wkt, bk, 1.0f,   Kh, nullptr);
  gemm_bf16_kernel<2><<<gg, 256, 0, stream>>>(Xv, Wvt, bv, 1.0f,   Vt, nullptr);

  const dim3 ag(SEQ / 128, BATCH * HEADS);
  attn_kernel<<<ag, 256, 0, stream>>>(Qb, Kh, Vt, mask, Cx);

  gemm_bf16_kernel<0><<<gg, 256, 0, stream>>>(Cx, Wot, bo, 1.0f, nullptr, (float*)d_out);
}
